// DSVDD_90297392431352
// MI455X (gfx1250) — matrix operand order repaired and checked by reading, not compiled, not run on hardware
//
#include <hip/hip_runtime.h>
#include <hip/hip_bf16.h>

// ---------------------------------------------------------------------------
// Problem constants (from reference)
// ---------------------------------------------------------------------------
#define NB     8
#define HH     56
#define WW     56
#define NPIX   25088      // NB*HH*WW
#define COUT   1792
#define KIN    1794       // DIM + 2 coord channels
#define KP     1824       // KIN padded to 57*32 for WMMA K-loop
#define MM     3136       // memory bank entries (= 49*64)
#define LDSROW 1800       // padded LDS row stride (halves) -> bank-conflict free

typedef __attribute__((ext_vector_type(16))) __bf16 v16bf;
typedef __attribute__((ext_vector_type(8)))  __bf16 v8bf;
typedef __attribute__((ext_vector_type(8)))  float  v8f;

// ---------------------------------------------------------------------------
// 3x3 avg pool, stride 1, pad 1, count_include_pad (divide by 9 always)
// ---------------------------------------------------------------------------
__global__ void pool3x3(const float* __restrict__ in, float* __restrict__ out,
                        int planes, int S) {
  int i = blockIdx.x * blockDim.x + threadIdx.x;
  int total = planes * S * S;
  if (i >= total) return;
  int x = i % S, y = (i / S) % S, p = i / (S * S);
  const float* pl = in + (size_t)p * S * S;
  int y0 = max(y - 1, 0), y1 = min(y + 1, S - 1);
  int x0 = max(x - 1, 0), x1 = min(x + 1, S - 1);
  float s = 0.f;
  for (int yy = y0; yy <= y1; ++yy)
    for (int xx = x0; xx <= x1; ++xx) s += pl[yy * S + xx];
  out[i] = s * (1.0f / 9.0f);
}

// ---------------------------------------------------------------------------
// conv_w fp32 [COUT][KIN] -> bf16 [COUT][KP] (zero padded K)
// ---------------------------------------------------------------------------
__global__ void prep_w(const float* __restrict__ w, __bf16* __restrict__ wt) {
  int i = blockIdx.x * blockDim.x + threadIdx.x;
  if (i >= COUT * KP) return;
  int o = i / KP, k = i % KP;
  float v = (k < KIN) ? w[(size_t)o * KIN + k] : 0.f;
  wt[i] = (__bf16)v;
}

// ---------------------------------------------------------------------------
// memory_bank fp32 [COUT][MM] -> bankT bf16 [MM][COUT] + column sumsq
// ---------------------------------------------------------------------------
__global__ void prep_bank(const float* __restrict__ mb, __bf16* __restrict__ bt,
                          float* __restrict__ c_sq) {
  int m = blockIdx.x * blockDim.x + threadIdx.x;
  if (m >= MM) return;
  float s = 0.f;
  for (int k = 0; k < COUT; ++k) {
    float v = mb[(size_t)k * MM + m];   // coalesced across lanes
    s += v * v;
    bt[(size_t)m * COUT + k] = (__bf16)v;
  }
  c_sq[m] = s;
}

// ---------------------------------------------------------------------------
// half-pixel bilinear from a pooled S x S plane up to 56 x 56
// ---------------------------------------------------------------------------
__device__ __forceinline__ float bilin(const float* __restrict__ pl, int S,
                                       float scale, int y, int x) {
  float sy = (y + 0.5f) * scale - 0.5f;
  float sx = (x + 0.5f) * scale - 0.5f;
  float fy = floorf(sy), fx = floorf(sx);
  float wy = sy - fy, wx = sx - fx;
  int y0 = (int)fy, x0 = (int)fx;
  int y1 = min(y0 + 1, S - 1), x1 = min(x0 + 1, S - 1);
  y0 = max(y0, 0); x0 = max(x0, 0);
  float v00 = pl[y0 * S + x0], v01 = pl[y0 * S + x1];
  float v10 = pl[y1 * S + x0], v11 = pl[y1 * S + x1];
  return v00 * (1.f - wy) * (1.f - wx) + v01 * (1.f - wy) * wx +
         v10 * wy * (1.f - wx) + v11 * wy * wx;
}

// ---------------------------------------------------------------------------
// Build descriptor-GEMM A matrix: bf16 [NPIX][KP], staged via LDS transpose
// so both global reads (pixel-fast) and A writes (channel-fast) coalesce.
// ---------------------------------------------------------------------------
__global__ __launch_bounds__(256) void build_input(
    const float* __restrict__ f1, const float* __restrict__ p2,
    const float* __restrict__ p3, __bf16* __restrict__ A) {
  __shared__ __bf16 tile[32][66];
  const int tid = threadIdx.x;
  const int cbase = blockIdx.x * 32;   // 57 tiles cover KP=1824
  const int nbase = blockIdx.y * 64;   // 392 tiles cover NPIX=25088

  for (int e = tid; e < 2048; e += 256) {
    int cl = e >> 6, nl = e & 63;
    int c = cbase + cl, n = nbase + nl;
    int b = n / 3136, rem = n % 3136, y = rem / 56, x = rem % 56;
    float val;
    if (c < 256) {                       // pool f1 on the fly
      const float* pl = f1 + (size_t)(b * 256 + c) * 3136;
      int y0 = max(y - 1, 0), y1 = min(y + 1, 55);
      int x0 = max(x - 1, 0), x1 = min(x + 1, 55);
      float s = 0.f;
      for (int yy = y0; yy <= y1; ++yy)
        for (int xx = x0; xx <= x1; ++xx) s += pl[yy * 56 + xx];
      val = s * (1.f / 9.f);
    } else if (c < 768) {
      val = bilin(p2 + (size_t)(b * 512 + (c - 256)) * 784, 28, 0.5f, y, x);
    } else if (c < 1792) {
      val = bilin(p3 + (size_t)(b * 1024 + (c - 768)) * 196, 14, 0.25f, y, x);
    } else if (c == 1792) {
      val = -1.f + (2.f / 55.f) * (float)x;
    } else if (c == 1793) {
      val = -1.f + (2.f / 55.f) * (float)y;
    } else {
      val = 0.f;                         // K padding
    }
    tile[cl][nl] = (__bf16)val;
  }
  __syncthreads();
  for (int e = tid; e < 2048; e += 256) {
    int cl = e & 31, nl = e >> 5;        // channel-fast -> coalesced stores
    A[(size_t)(nbase + nl) * KP + cbase + cl] = tile[cl][nl];
  }
}

// ---------------------------------------------------------------------------
// GEMM1: phi = A[NPIX x KP] * Wt^T + bias (NT, bf16 WMMA, fp32 accum).
// Wave owns 16x64 tile. Ping-pong buffers indexed by literal parity: two
// k-steps per iteration, no register copies, no in-loop bounds branches
// (KP = 57 steps = 28 pairs + peeled tail; in-loop prefetch stays in-bounds).
// ---------------------------------------------------------------------------
__global__ __launch_bounds__(256) void gemm_desc(
    const __bf16* __restrict__ A, const __bf16* __restrict__ Wt,
    const float* __restrict__ bias, __bf16* __restrict__ feats,
    float* __restrict__ f_sq) {
  const int wid = blockIdx.x * 8 + (threadIdx.x >> 5);
  const int lane = threadIdx.x & 31;
  const int hi = lane >> 4, lr = lane & 15;
  const int ct = wid % 28, rt = wid / 28;       // 28 col tiles of 64, 1568 row tiles
  const int rowbase = rt << 4, colbase = ct << 6;

  v8f acc[4] = {};
  const __bf16* arow = A + (size_t)(rowbase + lr) * KP + hi * 16;
  const __bf16* brow = Wt + (size_t)(colbase + lr) * KP + hi * 16;

  union AF { v16bf f; v8bf h[2]; };
  AF afb[2];
  v16bf bb[2][4];

  auto loadstep = [&](int p, int k) {
    afb[p].h[0] = *(const v8bf*)(arow + k);
    afb[p].h[1] = *(const v8bf*)(arow + k + 8);
#pragma unroll
    for (int j = 0; j < 4; ++j)
      bb[p][j] = *(const v16bf*)(brow + (size_t)j * 16 * KP + k);
  };
  auto mm = [&](int p) {
#pragma unroll
    for (int j = 0; j < 4; ++j)
      acc[j] = __builtin_amdgcn_wmma_f32_16x16x32_bf16(
          false, afb[p].f, false, bb[p][j], (short)0, acc[j], false, false);
  };

  loadstep(0, 0);
  int k = 0;
#pragma unroll 1
  for (int it = 0; it < 28; ++it, k += 64) {   // pairs; last prefetch k+64=1792 OK
    loadstep(1, k + 32);
    mm(0);
    loadstep(0, k + 64);
    mm(1);
  }
  mm(0);                                        // tail step k = 1792

  float ss[8];
#pragma unroll
  for (int v = 0; v < 8; ++v) ss[v] = 0.f;
#pragma unroll
  for (int j = 0; j < 4; ++j) {
    int col = colbase + j * 16 + lr;
    float bv = bias[col];
#pragma unroll
    for (int v = 0; v < 8; ++v) {
      float val = acc[j][v] + bv;                 // D row = v + 8*hi, col = lr
      feats[(size_t)(rowbase + v + 8 * hi) * COUT + col] = (__bf16)val;
      ss[v] += val * val;
    }
  }
#pragma unroll
  for (int v = 0; v < 8; ++v) {
    float s = ss[v];
    s += __shfl_xor(s, 1, 32);
    s += __shfl_xor(s, 2, 32);
    s += __shfl_xor(s, 4, 32);
    s += __shfl_xor(s, 8, 32);                    // reduce within 16-lane half
    if (lr == 0) atomicAdd(&f_sq[rowbase + v + 8 * hi], s);
  }
}

// keep 3 smallest, sorted ascending
__device__ __forceinline__ void ins3(float v, float& a, float& b, float& c) {
  if (v < c) {
    if (v < b) {
      c = b;
      if (v < a) { b = a; a = v; } else { b = v; }
    } else {
      c = v;
    }
  }
}

// ---------------------------------------------------------------------------
// GEMM2 + fused top-3 softmin: dist^2 = f_sq + c_sq - 2 * feats.bankT^T.
// A tile brought into LDS with GLOBAL_LOAD_ASYNC_TO_LDS_B128 (ASYNCcnt path);
// 8 waves stripe the 49 column blocks of 64; ping-pong k-loop
// (COUT = 56 steps = 27 pairs + peeled load/wmma/wmma tail).
// ---------------------------------------------------------------------------
__global__ __launch_bounds__(256) void dist_topk(
    const __bf16* __restrict__ feats, const float* __restrict__ f_sq,
    const __bf16* __restrict__ bankT, const float* __restrict__ c_sq,
    float* __restrict__ out) {
  __shared__ __bf16 As[16 * LDSROW];          // 57.6 KB, padded stride
  __shared__ float red[16][25];               // 8 waves * 3 + pad
  const int tid = threadIdx.x;
  const int wv = tid >> 5, lane = tid & 31, hi = lane >> 4, lr = lane & 15;
  const int rowbase = blockIdx.x << 4;

  // async copy of the 16x1792 bf16 tile into LDS (padded row stride).
  // Each lane moves 16B per op: LDS dest offset rides in the first VGPR,
  // global address in the second (VGLOBAL async encoding, tracked by ASYNCcnt).
  for (int i = tid; i < 16 * (COUT / 8); i += 256) {
    int r = i / (COUT / 8), cch = i % (COUT / 8);
    unsigned int ldsoff =
        (unsigned int)(uintptr_t)(As + r * LDSROW + cch * 8);
    const __bf16* g = feats + (size_t)(rowbase + r) * COUT + cch * 8;
    asm volatile("global_load_async_to_lds_b128 %0, %1, off"
                 :: "v"(ldsoff), "v"(g) : "memory");
  }
  asm volatile("s_wait_asynccnt 0x0" ::: "memory");
  __syncthreads();

  float fr[8];
#pragma unroll
  for (int v = 0; v < 8; ++v) fr[v] = f_sq[rowbase + v + 8 * hi];

  float t0[8], t1[8], t2[8];
#pragma unroll
  for (int v = 0; v < 8; ++v) { t0[v] = 3.0e38f; t1[v] = 3.0e38f; t2[v] = 3.0e38f; }

  const __bf16* arow = As + lr * LDSROW + hi * 16;
  union AF { v16bf f; v8bf h[2]; };

  for (int cb = wv; cb < (MM / 64); cb += 8) {
    const int colbase = cb << 6;
    const __bf16* brow = bankT + (size_t)(colbase + lr) * COUT + hi * 16;

    {  // warm the next column block this wave will touch (global_prefetch_b8)
      int cbn = cb + 8;
      if (cbn < (MM / 64))
        __builtin_prefetch(bankT + (size_t)((cbn << 6) + lane) * COUT, 0, 0);
    }

    v8f acc[4] = {};
    AF afb[2];
    v16bf bb[2][4];

    auto loadstep = [&](int p, int k) {
      afb[p].h[0] = *(const v8bf*)(arow + k);        // ds_load_b128
      afb[p].h[1] = *(const v8bf*)(arow + k + 8);
#pragma unroll
      for (int j = 0; j < 4; ++j)
        bb[p][j] = *(const v16bf*)(brow + (size_t)j * 16 * COUT + k);
    };
    auto mm = [&](int p) {
#pragma unroll
      for (int j = 0; j < 4; ++j)
        acc[j] = __builtin_amdgcn_wmma_f32_16x16x32_bf16(
            false, afb[p].f, false, bb[p][j], (short)0, acc[j], false, false);
    };

    loadstep(0, 0);
    int k = 0;
#pragma unroll 1
    for (int it = 0; it < 27; ++it, k += 64) {  // last prefetch k+64=1728 OK
      loadstep(1, k + 32);
      mm(0);
      loadstep(0, k + 64);
      mm(1);
    }
    loadstep(1, 1760);                           // peeled tail: steps 1728, 1760
    mm(0);
    mm(1);

#pragma unroll
    for (int j = 0; j < 4; ++j) {
      int col = colbase + j * 16 + lr;
      float cs = c_sq[col];
#pragma unroll
      for (int v = 0; v < 8; ++v) {
        float d2 = fr[v] + cs - 2.0f * acc[j][v];
        ins3(d2, t0[v], t1[v], t2[v]);
      }
    }
  }

  // butterfly merge of top-3 sets within each 16-lane half (cols of a row)
#pragma unroll
  for (int m = 1; m <= 8; m <<= 1) {
#pragma unroll
    for (int v = 0; v < 8; ++v) {
      float o0 = __shfl_xor(t0[v], m, 32);
      float o1 = __shfl_xor(t1[v], m, 32);
      float o2 = __shfl_xor(t2[v], m, 32);
      ins3(o0, t0[v], t1[v], t2[v]);
      ins3(o1, t0[v], t1[v], t2[v]);
      ins3(o2, t0[v], t1[v], t2[v]);
    }
  }
  if (lr == 0) {
#pragma unroll
    for (int v = 0; v < 8; ++v) {
      int r = v + 8 * hi;
      red[r][wv * 3 + 0] = t0[v];
      red[r][wv * 3 + 1] = t1[v];
      red[r][wv * 3 + 2] = t2[v];
    }
  }
  __syncthreads();

  if (tid < 16) {  // merge across waves, softmin score
    float a0 = 3.0e38f, a1 = 3.0e38f, a2 = 3.0e38f;
    for (int i = 0; i < 24; ++i) ins3(red[tid][i], a0, a1, a2);
    float d0 = sqrtf(fmaxf(a0, 0.f));
    float d1 = sqrtf(fmaxf(a1, 0.f));
    float d2 = sqrtf(fmaxf(a2, 0.f));
    // softmax(-d)[0] * d0, stabilized around d0 (the max of -d)
    float denom = 1.0f + __expf(d0 - d1) + __expf(d0 - d2);
    out[rowbase + tid] = d0 / denom;
  }
}

// ---------------------------------------------------------------------------
extern "C" void kernel_launch(void* const* d_in, const int* in_sizes, int n_in,
                              void* d_out, int out_size, void* d_ws, size_t ws_size,
                              hipStream_t stream) {
  const float* f1     = (const float*)d_in[0];  // [8,256,56,56]
  const float* f2     = (const float*)d_in[1];  // [8,512,28,28]
  const float* f3     = (const float*)d_in[2];  // [8,1024,14,14]
  const float* conv_w = (const float*)d_in[3];  // [1792,1794]
  const float* conv_b = (const float*)d_in[4];  // [1792]
  const float* mbank  = (const float*)d_in[5];  // [1792,3136]
  float* out = (float*)d_out;                   // [8,1,56,56] fp32

  char* ws = (char*)d_ws;
  size_t off = 0;
  auto alloc = [&](size_t bytes) -> void* {
    void* p = ws + off;
    off = (off + bytes + 255) & ~(size_t)255;
    return p;
  };
  float*  p2    = (float*)alloc(sizeof(float) * (size_t)NB * 512 * 28 * 28);
  float*  p3    = (float*)alloc(sizeof(float) * (size_t)NB * 1024 * 14 * 14);
  __bf16* A     = (__bf16*)alloc(2ull * NPIX * KP);
  __bf16* Wt    = (__bf16*)alloc(2ull * COUT * KP);
  __bf16* feats = (__bf16*)alloc(2ull * NPIX * COUT);
  float*  f_sq  = (float*)alloc(sizeof(float) * NPIX);
  __bf16* bankT = (__bf16*)alloc(2ull * MM * COUT);
  float*  c_sq  = (float*)alloc(sizeof(float) * MM);

  hipMemsetAsync(f_sq, 0, sizeof(float) * NPIX, stream);

  pool3x3<<<(NB * 512 * 28 * 28 + 255) / 256, 256, 0, stream>>>(f2, p2, NB * 512, 28);
  pool3x3<<<(NB * 1024 * 14 * 14 + 255) / 256, 256, 0, stream>>>(f3, p3, NB * 1024, 14);
  prep_w<<<(COUT * KP + 255) / 256, 256, 0, stream>>>(conv_w, Wt);
  prep_bank<<<(MM + 255) / 256, 256, 0, stream>>>(mbank, bankT, c_sq);
  build_input<<<dim3(KP / 32, NPIX / 64), 256, 0, stream>>>(f1, p2, p3, A);
  gemm_desc<<<(NPIX / 16) * (COUT / 64) / 8, 256, 0, stream>>>(A, Wt, conv_b, feats, f_sq);
  dist_topk<<<NPIX / 16, 256, 0, stream>>>(feats, f_sq, bankT, c_sq, out);
}
